// AttentionBlock_31181462569020
// MI455X (gfx1250) — matrix operand order repaired and checked by reading, not compiled, not run on hardware
//
#include <hip/hip_runtime.h>

typedef __bf16 bf16_t;
typedef __attribute__((ext_vector_type(16))) __bf16 v16bf;
typedef __attribute__((ext_vector_type(8)))  float  v8f;
typedef __attribute__((ext_vector_type(4)))  unsigned int u32x4;
typedef __attribute__((ext_vector_type(4)))  float  f32x4;

// ---------------- workspace layout (bytes) ----------------
#define OFF_STATS 0                      // 64 floats: (mu, rstd) x 32 groups
#define OFF_XNT   (4096)                 // bf16 [2][4096][256]   (4 MB)
#define OFF_WQKV  (OFF_XNT  + 4194304)   // bf16 [768][256]       (384 KB)
#define OFF_WOUT  (OFF_WQKV + 393216)    // bf16 [256][256]       (128 KB)
#define OFF_QT    (OFF_WOUT + 131072)    // bf16 [2][4][4096][64] (4 MB)  q*scale
#define OFF_KT    (OFF_QT   + 4194304)   // bf16 [2][4][4096][64] (4 MB)  k*scale
#define OFF_V     (OFF_KT   + 4194304)   // bf16 [2][4][64][4096] (4 MB)
#define OFF_AT    (OFF_V    + 4194304)   // bf16 [2][4096][256]   (4 MB)

#define LDSPAD 72   // padded row stride (elements): 144 B -> conflict-free frags

__device__ inline v8f v8f_zero() {
  v8f z;
#pragma unroll
  for (int i = 0; i < 8; ++i) z[i] = 0.f;
  return z;
}

__device__ inline v8f wmma_bf16(v16bf a, v16bf b, v8f c) {
  return __builtin_amdgcn_wmma_f32_16x16x32_bf16(false, a, false, b, (short)0, c,
                                                 false, false);
}

// A-matrix 16x32 bf16 fragment. src is [row][k], k contiguous, row stride
// `rs` elements. ISA layout: lanes 0-15 hold K 0-7 / 16-23, lanes 16-31 hold
// K 8-15 / 24-31, M = lane&15.
__device__ inline v16bf load_frag_a(const bf16_t* base, int rs, int lane) {
  int m  = lane & 15;
  int k0 = (lane & 16) ? 8 : 0;
  const bf16_t* p = base + m * rs + k0;
  union { v16bf v; u32x4 q[2]; } u;
  u.q[0] = *(const u32x4*)(p);
  u.q[1] = *(const u32x4*)(p + 16);
  return u.v;
}

// B-matrix 32x16 bf16 fragment. src is [n][k], k contiguous, row stride `rs`.
// ISA layout: lanes 0-15 hold K 0-7 / 16-23, lanes 16-31 hold K 8-15 / 24-31, N = lane&15.
__device__ inline v16bf load_frag_b(const bf16_t* base, int rs, int lane) {
  int n  = lane & 15;
  int k0 = (lane & 16) ? 8 : 0;
  const bf16_t* p = base + n * rs + k0;
  union { v16bf v; u32x4 q[2]; } u;
  u.q[0] = *(const u32x4*)(p);
  u.q[1] = *(const u32x4*)(p + 16);
  return u.v;
}

// ---------------- 1) GroupNorm statistics ----------------
__global__ void gn_stats_kernel(const float* __restrict__ x, float* __restrict__ stats) {
  int b = blockIdx.x;                       // n*16 + g, 32 blocks
  const f32x4* p = (const f32x4*)(x + (size_t)b * 65536);
  float s = 0.f, ss = 0.f;
  for (int i = threadIdx.x; i < 16384; i += 256) {
    f32x4 v = p[i];
#pragma unroll
    for (int j = 0; j < 4; ++j) { s += v[j]; ss += v[j] * v[j]; }
  }
#pragma unroll
  for (int off = 16; off; off >>= 1) {
    s  += __shfl_down(s, off);
    ss += __shfl_down(ss, off);
  }
  __shared__ float red[2][8];
  int lane = threadIdx.x & 31, wave = threadIdx.x >> 5;
  if (lane == 0) { red[0][wave] = s; red[1][wave] = ss; }
  __syncthreads();
  if (threadIdx.x == 0) {
    float S = 0.f, SS = 0.f;
#pragma unroll
    for (int w = 0; w < 8; ++w) { S += red[0][w]; SS += red[1][w]; }
    float mu  = S * (1.f / 65536.f);
    float var = SS * (1.f / 65536.f) - mu * mu;
    stats[b * 2]     = mu;
    stats[b * 2 + 1] = rsqrtf(var + 1e-6f);
  }
}

// ---------------- 2) GroupNorm apply -> xnT bf16 [n][l][c] ----------------
__global__ void gn_apply_kernel(const float* __restrict__ x,
                                const float* __restrict__ gsc,
                                const float* __restrict__ gbi,
                                const float* __restrict__ stats,
                                bf16_t* __restrict__ xnT) {
  int idx = blockIdx.x * 256 + threadIdx.x;   // 2*256*4096 threads
  int l = idx & 4095;
  int c = (idx >> 12) & 255;
  int n = idx >> 20;
  int g = c >> 4;
  float mu = stats[(n * 16 + g) * 2];
  float rs = stats[(n * 16 + g) * 2 + 1];
  float v = (x[idx] - mu) * rs * gsc[c] + gbi[c];
  xnT[((size_t)n << 20) + ((size_t)l << 8) + c] = (bf16_t)v;
}

// ---------------- 3) weight casts ----------------
__global__ void cast_w_kernel(const float* __restrict__ wq, const float* __restrict__ wo,
                              bf16_t* __restrict__ wqb, bf16_t* __restrict__ wob) {
  int idx = blockIdx.x * 256 + threadIdx.x;   // 262144 threads
  if (idx < 196608) wqb[idx] = (bf16_t)wq[idx];
  else              wob[idx - 196608] = (bf16_t)wo[idx - 196608];
}

// ---------------- 4) QKV GEMM: qkv[n,o,l] = w_qkv[o,:] . xn[n,:,l] ----------
// One wave computes a 32(o) x 64(l) tile (2 A tiles share every B fragment).
__global__ void qkv_gemm_kernel(const bf16_t* __restrict__ wq,
                                const bf16_t* __restrict__ xnT,
                                const float* __restrict__ b_qkv,
                                bf16_t* __restrict__ qT,
                                bf16_t* __restrict__ kT,
                                bf16_t* __restrict__ vv) {
  int lane = threadIdx.x & 31, wave = threadIdx.x >> 5;
  int tile = blockIdx.x * 8 + wave;           // 3072 tiles = 2 * 24 * 64
  int n  = tile / 1536;
  int r0 = tile % 1536;
  int op = r0 >> 6;                           // 0..23 (pair of o tiles)
  int lq = r0 & 63;                           // 0..63 (l chunk of 64)
  const bf16_t* A = wq + (size_t)op * 32 * 256;
  const bf16_t* B = xnT + ((size_t)n << 20) + (size_t)lq * 64 * 256;
  v8f acc[2][4];
#pragma unroll
  for (int p = 0; p < 2; ++p)
#pragma unroll
    for (int j = 0; j < 4; ++j) acc[p][j] = v8f_zero();
#pragma unroll
  for (int kc = 0; kc < 8; ++kc) {
    v16bf a0 = load_frag_a(A + kc * 32, 256, lane);
    v16bf a1 = load_frag_a(A + 16 * 256 + kc * 32, 256, lane);
#pragma unroll
    for (int j = 0; j < 4; ++j) {
      v16bf b = load_frag_b(B + (size_t)j * 16 * 256 + kc * 32, 256, lane);
      acc[0][j] = wmma_bf16(a0, b, acc[0][j]);
      acc[1][j] = wmma_bf16(a1, b, acc[1][j]);
    }
  }
  const float scale = 0.3535533906f;          // 1/sqrt(sqrt(64))
  int rb = (lane & 16) ? 8 : 0;
#pragma unroll
  for (int p = 0; p < 2; ++p) {
#pragma unroll
    for (int j = 0; j < 4; ++j) {
      int l = lq * 64 + j * 16 + (lane & 15);
#pragma unroll
      for (int r = 0; r < 8; ++r) {
        int o = (op * 2 + p) * 16 + rb + r;
        float val = acc[p][j][r] + b_qkv[o];
        int part = o >> 8, rr = o & 255, h = rr >> 6, cc = rr & 63;
        size_t nh = (size_t)n * 4 + h;
        if (part == 0)      qT[(nh * 4096 + l) * 64 + cc] = (bf16_t)(val * scale);
        else if (part == 1) kT[(nh * 4096 + l) * 64 + cc] = (bf16_t)(val * scale);
        else                vv[(nh * 64 + cc) * 4096 + l] = (bf16_t)val;
      }
    }
  }
}

// ---------------- 5) Flash attention ----------------
// Block = (n, h, tblk): 128 t rows, 8 waves x 16 rows. s swept in 64-blocks.
// K/V tiles for each s-block are staged in LDS once and shared by all waves.
__global__ void flash_attn_kernel(const bf16_t* __restrict__ qT,
                                  const bf16_t* __restrict__ kT,
                                  const bf16_t* __restrict__ vv,
                                  bf16_t* __restrict__ aT) {
  int lane = threadIdx.x & 31, wave = threadIdx.x >> 5;
  int tid  = threadIdx.x;
  int blk = blockIdx.x;                       // 256 = 2 * 4 * 32
  int n = blk >> 7, h = (blk >> 5) & 3, tblk = blk & 31;
  int tw = tblk * 128 + wave * 16;
  size_t nh = (size_t)n * 4 + h;
  const bf16_t* qb = qT + (nh * 4096 + tw) * 64;
  const bf16_t* kb = kT + nh * 4096 * 64;
  const bf16_t* vb = vv + nh * 64 * 4096;

  __shared__ __align__(16) bf16_t ktile[64 * LDSPAD];     // [s_local][c]
  __shared__ __align__(16) bf16_t vtile[64 * LDSPAD];     // [c][s_local]
  __shared__ __align__(16) bf16_t pbuf[8][16 * LDSPAD];   // per-wave P tile
  bf16_t* pw = pbuf[wave];

  v16bf aq0 = load_frag_a(qb, 64, lane);          // c 0..31
  v16bf aq1 = load_frag_a(qb + 32, 64, lane);     // c 32..63

  v8f o0 = v8f_zero(), o1 = v8f_zero(), o2 = v8f_zero(), o3 = v8f_zero();
  float m[8], lsum[8];
#pragma unroll
  for (int r = 0; r < 8; ++r) { m[r] = -1e30f; lsum[r] = 0.f; }

  int col = lane & 15;
  int rb  = (lane & 16) ? 8 : 0;

  for (int sb = 0; sb < 64; ++sb) {
    int s0 = sb * 64;
    // ---- cooperative K/V tile staging: 2 x 8KB, b128 in / b128 to LDS ----
#pragma unroll
    for (int i = 0; i < 2; ++i) {
      int idx  = tid + i * 256;                 // 0..511 chunks of 8 bf16
      int row  = idx >> 3;
      int seg  = (idx & 7) * 8;
      *(u32x4*)(ktile + row * LDSPAD + seg) =
          *(const u32x4*)(kb + (size_t)(s0 + row) * 64 + seg);
      *(u32x4*)(vtile + row * LDSPAD + seg) =
          *(const u32x4*)(vb + (size_t)row * 4096 + s0 + seg);
    }
    __syncthreads();

    // ---- S = (q*scale).(k*scale)^T : 16t x 64s, fragments from LDS ----
    v8f S[4];
#pragma unroll
    for (int j = 0; j < 4; ++j) {
      v8f s = v8f_zero();
      const bf16_t* krow = ktile + (size_t)(j * 16) * LDSPAD;
      s = wmma_bf16(aq0, load_frag_b(krow, LDSPAD, lane), s);
      s = wmma_bf16(aq1, load_frag_b(krow + 32, LDSPAD, lane), s);
      S[j] = s;
    }
    // ---- online softmax; shfl_xor masks < 16 keep lane groups separate ----
#pragma unroll
    for (int r = 0; r < 8; ++r) {
      float rm = fmaxf(fmaxf(S[0][r], S[1][r]), fmaxf(S[2][r], S[3][r]));
      rm = fmaxf(rm, __shfl_xor(rm, 1));
      rm = fmaxf(rm, __shfl_xor(rm, 2));
      rm = fmaxf(rm, __shfl_xor(rm, 4));
      rm = fmaxf(rm, __shfl_xor(rm, 8));
      float mn = fmaxf(m[r], rm);
      float sc = __expf(m[r] - mn);
      m[r] = mn;
      float p0 = __expf(S[0][r] - mn), p1 = __expf(S[1][r] - mn);
      float p2 = __expf(S[2][r] - mn), p3 = __expf(S[3][r] - mn);
      S[0][r] = p0; S[1][r] = p1; S[2][r] = p2; S[3][r] = p3;
      float rs = p0 + p1 + p2 + p3;
      rs += __shfl_xor(rs, 1);
      rs += __shfl_xor(rs, 2);
      rs += __shfl_xor(rs, 4);
      rs += __shfl_xor(rs, 8);
      lsum[r] = lsum[r] * sc + rs;
      o0[r] *= sc; o1[r] *= sc; o2[r] *= sc; o3[r] *= sc;
    }
    // ---- C-layout -> A-layout bounce of P through this wave's LDS slice ----
    // (within-wave DS ops are in-order; no block barrier needed here)
#pragma unroll
    for (int r = 0; r < 8; ++r) {
      pw[(rb + r) * LDSPAD + 0  + col] = (bf16_t)S[0][r];
      pw[(rb + r) * LDSPAD + 16 + col] = (bf16_t)S[1][r];
      pw[(rb + r) * LDSPAD + 32 + col] = (bf16_t)S[2][r];
      pw[(rb + r) * LDSPAD + 48 + col] = (bf16_t)S[3][r];
    }
    v16bf ap0 = load_frag_a(pw, LDSPAD, lane);        // s0+0..31
    v16bf ap1 = load_frag_a(pw + 32, LDSPAD, lane);   // s0+32..63
    o0 = wmma_bf16(ap0, load_frag_b(vtile + (size_t)0  * LDSPAD,      LDSPAD, lane), o0);
    o0 = wmma_bf16(ap1, load_frag_b(vtile + (size_t)0  * LDSPAD + 32, LDSPAD, lane), o0);
    o1 = wmma_bf16(ap0, load_frag_b(vtile + (size_t)16 * LDSPAD,      LDSPAD, lane), o1);
    o1 = wmma_bf16(ap1, load_frag_b(vtile + (size_t)16 * LDSPAD + 32, LDSPAD, lane), o1);
    o2 = wmma_bf16(ap0, load_frag_b(vtile + (size_t)32 * LDSPAD,      LDSPAD, lane), o2);
    o2 = wmma_bf16(ap1, load_frag_b(vtile + (size_t)32 * LDSPAD + 32, LDSPAD, lane), o2);
    o3 = wmma_bf16(ap0, load_frag_b(vtile + (size_t)48 * LDSPAD,      LDSPAD, lane), o3);
    o3 = wmma_bf16(ap1, load_frag_b(vtile + (size_t)48 * LDSPAD + 32, LDSPAD, lane), o3);
    __syncthreads();
  }
  // ---- normalize and store aT[n][t][h*64 + c] ----
#pragma unroll
  for (int r = 0; r < 8; ++r) {
    float inv = 1.f / lsum[r];
    int t = tw + rb + r;
    size_t row = ((size_t)n * 4096 + t) * 256 + h * 64;
    aT[row + 0  + col] = (bf16_t)(o0[r] * inv);
    aT[row + 16 + col] = (bf16_t)(o1[r] * inv);
    aT[row + 32 + col] = (bf16_t)(o2[r] * inv);
    aT[row + 48 + col] = (bf16_t)(o3[r] * inv);
  }
}

// ---------------- 6) out-proj + bias + residual ----------------
__global__ void out_proj_kernel(const bf16_t* __restrict__ wo,
                                const bf16_t* __restrict__ aT,
                                const float* __restrict__ b_out,
                                const float* __restrict__ x,
                                float* __restrict__ out) {
  int lane = threadIdx.x & 31, wave = threadIdx.x >> 5;
  int tile = blockIdx.x * 8 + wave;           // 1024 tiles = 2 * 8 * 64
  int n  = tile >> 9;
  int r0 = tile & 511;
  int op = r0 >> 6;                           // 0..7 (pair of o tiles)
  int lq = r0 & 63;                           // 0..63
  const bf16_t* A = wo + (size_t)op * 32 * 256;
  const bf16_t* B = aT + ((size_t)n << 20) + (size_t)lq * 64 * 256;
  v8f acc[2][4];
#pragma unroll
  for (int p = 0; p < 2; ++p)
#pragma unroll
    for (int j = 0; j < 4; ++j) acc[p][j] = v8f_zero();
#pragma unroll
  for (int kc = 0; kc < 8; ++kc) {
    v16bf a0 = load_frag_a(A + kc * 32, 256, lane);
    v16bf a1 = load_frag_a(A + 16 * 256 + kc * 32, 256, lane);
#pragma unroll
    for (int j = 0; j < 4; ++j) {
      v16bf b = load_frag_b(B + (size_t)j * 16 * 256 + kc * 32, 256, lane);
      acc[0][j] = wmma_bf16(a0, b, acc[0][j]);
      acc[1][j] = wmma_bf16(a1, b, acc[1][j]);
    }
  }
  int rb = (lane & 16) ? 8 : 0;
#pragma unroll
  for (int p = 0; p < 2; ++p) {
#pragma unroll
    for (int j = 0; j < 4; ++j) {
      int l = lq * 64 + j * 16 + (lane & 15);
#pragma unroll
      for (int r = 0; r < 8; ++r) {
        int o = (op * 2 + p) * 16 + rb + r;
        size_t idx = ((size_t)n * 256 + o) * 4096 + l;
        out[idx] = x[idx] + acc[p][j][r] + b_out[o];
      }
    }
  }
}

// ---------------- launch ----------------
extern "C" void kernel_launch(void* const* d_in, const int* in_sizes, int n_in,
                              void* d_out, int out_size, void* d_ws, size_t ws_size,
                              hipStream_t stream) {
  const float* x      = (const float*)d_in[0];
  const float* gsc    = (const float*)d_in[1];
  const float* gbi    = (const float*)d_in[2];
  const float* w_qkv  = (const float*)d_in[3];
  const float* b_qkv  = (const float*)d_in[4];
  const float* w_out  = (const float*)d_in[5];
  const float* b_out  = (const float*)d_in[6];

  char* ws = (char*)d_ws;
  float*  stats = (float*)(ws + OFF_STATS);
  bf16_t* xnT   = (bf16_t*)(ws + OFF_XNT);
  bf16_t* wqb   = (bf16_t*)(ws + OFF_WQKV);
  bf16_t* wob   = (bf16_t*)(ws + OFF_WOUT);
  bf16_t* qT    = (bf16_t*)(ws + OFF_QT);
  bf16_t* kT    = (bf16_t*)(ws + OFF_KT);
  bf16_t* vv    = (bf16_t*)(ws + OFF_V);
  bf16_t* aT    = (bf16_t*)(ws + OFF_AT);

  gn_stats_kernel<<<32, 256, 0, stream>>>(x, stats);
  gn_apply_kernel<<<8192, 256, 0, stream>>>(x, gsc, gbi, stats, xnT);
  cast_w_kernel<<<1024, 256, 0, stream>>>(w_qkv, w_out, wqb, wob);
  qkv_gemm_kernel<<<384, 256, 0, stream>>>(wqb, xnT, b_qkv, qT, kT, vv);
  flash_attn_kernel<<<256, 256, 0, stream>>>(qT, kT, vv, aT);
  out_proj_kernel<<<128, 256, 0, stream>>>(wob, aT, b_out, x, (float*)d_out);
}
